// GraphConvolutionLayer_23098334118224
// MI455X (gfx1250) — matrix operand order repaired and checked by reading, not compiled, not run on hardware
//
#include <hip/hip_runtime.h>
#include <hip/hip_bf16.h>

// ---------------------------------------------------------------------------
// GraphConvolutionLayer on MI455X (gfx1250, wave32, WMMA)
//   out = diag(1/(adj.sum(1)+beta)) @ (adj @ x + beta*x) @ kernel + bias
// N = 8192, F_IN = F_OUT = 128
//
// adj (256MB fp32) is read exactly once (HBM floor ~11us @ 23.3TB/s),
// converted to bf16 in-register and fed to v_wmma_f32_16x16x32_bf16 with fp32
// accumulate; the rowsum is fused in fp32 on the original values. Split-K with
// deterministic per-chunk partial buffers (no float atomics); split factor is
// a template parameter (compile-time trip counts -> unrolled, deeply pipelined
// loop), selected at launch from ws_size (prefer 8 for HBM saturation).
// Second (tiny) GEMM done with WMMA in the epilogue. adj stream prefetched
// 4KB/row ahead via global_prefetch_b8.
// ---------------------------------------------------------------------------

typedef __bf16 bf16_t;
typedef __attribute__((ext_vector_type(16))) __bf16 v16bf;
typedef __attribute__((ext_vector_type(8)))  float  v8f;
typedef __attribute__((ext_vector_type(4)))  float  v4f;

static constexpr int kN      = 8192;
static constexpr int kFin    = 128;
static constexpr int kFout   = 128;
static constexpr int kStrips = kN / 16;     // 512 row strips

// Fixed workspace prefix (bytes): xT then wT, then split-sized partials.
static constexpr size_t kOffXT  = 0;                                 // bf16 xT[128][8192]
static constexpr size_t kOffWT  = kOffXT + (size_t)kFin * kN * 2;    // bf16 wT[128][128]
static constexpr size_t kOffVar = kOffWT + (size_t)kFout * kFin * 2; // partials start here

__device__ __forceinline__ bf16_t to_bf16(float f) { return (bf16_t)f; }

__device__ __forceinline__ v8f wmma_bf16(v16bf a, v16bf b, v8f c) {
  return __builtin_amdgcn_wmma_f32_16x16x32_bf16(
      /*neg_a=*/false, a, /*neg_b=*/false, b,
      /*c_mod=*/(short)0, c, /*reuse_a=*/false, /*reuse_b=*/false);
}

// ---------------------------------------------------------------------------
// Prep: build xT (bf16, [F_in][N]) and wT (bf16, [F_out][F_in]) so WMMA
// B-fragments are 32B-contiguous per lane.
// ---------------------------------------------------------------------------
__global__ __launch_bounds__(256) void gcn_prep(const float* __restrict__ x,
                                                const float* __restrict__ w,
                                                bf16_t* __restrict__ xT,
                                                bf16_t* __restrict__ wT) {
  const int idx = blockIdx.x * blockDim.x + threadIdx.x;
  if (idx < kFin * kN) {
    const int n = idx / kN;      // feature
    const int k = idx % kN;      // node
    xT[idx] = to_bf16(x[(size_t)k * kFin + n]);
  }
  if (idx < kFout * kFin) {
    const int n = idx / kFin;    // out feature
    const int k = idx % kFin;    // in feature
    wT[idx] = to_bf16(w[k * kFout + n]);
  }
}

// ---------------------------------------------------------------------------
// Main pass: per wave = one 16-row strip x one K-chunk (compile-time SPLIT).
//   acc[16x128] += bf16(adjTile) @ bf16(xT tile)   (fp32 accumulate, WMMA)
//   rowsum fp32 fused on the original adj values.
// ---------------------------------------------------------------------------
template <int SPLIT>
__global__ __launch_bounds__(256) void gcn_main(const float* __restrict__ adj,
                                                const bf16_t* __restrict__ xT,
                                                float* __restrict__ pagg,
                                                float* __restrict__ prs) {
  constexpr int kKC = kN / SPLIT;           // K per chunk (compile-time)
  const int wave  = (blockIdx.x * blockDim.x + threadIdx.x) >> 5;
  const int lane  = threadIdx.x & 31;
  const int strip = wave & (kStrips - 1);
  const int chunk = wave >> 9;              // kStrips == 512
  const int l15   = lane & 15;
  const int lhi   = lane >> 4;              // 0 or 1 (half-wave)
  const int r0    = strip * 16;
  const int row   = r0 + l15;               // A-matrix: lane holds one row
  const float* arow = adj + (size_t)row * kN;
  const int k0 = chunk * kKC;

  v8f acc[8];
#pragma unroll
  for (int i = 0; i < 8; ++i)
#pragma unroll
    for (int e = 0; e < 8; ++e) acc[i][e] = 0.0f;
  float rs = 0.0f;

#pragma unroll 2
  for (int it = 0; it < kKC / 32; ++it) {
    const int kb = k0 + it * 32;
    // --- A fragment: 16x32 bf16. lanes 0-15 -> K 0..15,
    //     lanes 16-31 -> K 16..31 of the same rows.
    const int kA = kb + lhi * 16;
    const v4f a0 = *(const v4f*)(arow + kA);
    const v4f a1 = *(const v4f*)(arow + kA + 4);
    const v4f a2 = *(const v4f*)(arow + kA + 8);
    const v4f a3 = *(const v4f*)(arow + kA + 12);

    // stream prefetch ~32 iterations (4KB/row) ahead; speculative, safely
    // dropped past end of the allocation
    __builtin_prefetch(arow + kA + 1024, 0, 1);

    // fused rowsum in fp32 (lanes l and l+16 cover complementary K halves)
    rs += (a0.x + a0.y + a0.z + a0.w) + (a1.x + a1.y + a1.z + a1.w) +
          (a2.x + a2.y + a2.z + a2.w) + (a3.x + a3.y + a3.z + a3.w);

    v16bf af;
    af[0] = to_bf16(a0.x);  af[1] = to_bf16(a0.y);
    af[2] = to_bf16(a0.z);  af[3] = to_bf16(a0.w);
    af[4] = to_bf16(a1.x);  af[5] = to_bf16(a1.y);
    af[6] = to_bf16(a1.z);  af[7] = to_bf16(a1.w);
    af[8]  = to_bf16(a2.x); af[9]  = to_bf16(a2.y);
    af[10] = to_bf16(a2.z); af[11] = to_bf16(a2.w);
    af[12] = to_bf16(a3.x); af[13] = to_bf16(a3.y);
    af[14] = to_bf16(a3.z); af[15] = to_bf16(a3.w);

    // --- B fragments: 32x16 bf16 per N-tile. Lane = column, 16 contiguous K
    //     values (lanes 0-15: K 0-15, lanes 16-31: K 16-31). xT is [F][N] so
    //     these are 32B-contiguous loads (L2-resident, 2MB total).
    const bf16_t* bbase = xT + (size_t)l15 * kN + (kb + lhi * 16);
#pragma unroll
    for (int nn = 0; nn < 8; ++nn) {
      const v16bf bfrag = *(const v16bf*)(bbase + (size_t)(nn * 16) * kN);
      acc[nn] = wmma_bf16(af, bfrag, acc[nn]);
    }
  }

  // combine rowsum halves across half-waves, store per-chunk partial
  rs += __shfl_xor(rs, 16, 32);
  if (lhi == 0) prs[chunk * kN + row] = rs;

  // store partial agg tile. C layout: VGPR j, lanes<16 -> M=j, lanes>=16 -> M=j+8
  float* pa = pagg + (size_t)chunk * kN * kFin;
#pragma unroll
  for (int nn = 0; nn < 8; ++nn) {
    const int col = nn * 16 + l15;
#pragma unroll
    for (int j = 0; j < 8; ++j)
      pa[(size_t)(r0 + j + 8 * lhi) * kFin + col] = acc[nn][j];
  }
}

// ---------------------------------------------------------------------------
// Epilogue: per wave = one 16-row strip (compile-time SPLIT).
//   agg = sum(partials) + beta*x ; scaled by 1/(rowsum+beta)
//   out = bf16(agg) @ bf16(kernel) + bias   (WMMA, K=128)
// ---------------------------------------------------------------------------
template <int SPLIT>
__global__ __launch_bounds__(256) void gcn_epilogue(const float* __restrict__ x,
                                                    const float* __restrict__ bias,
                                                    const float* __restrict__ beta,
                                                    const bf16_t* __restrict__ wT,
                                                    const float* __restrict__ pagg,
                                                    const float* __restrict__ prs,
                                                    float* __restrict__ out) {
  const int wave = (blockIdx.x * blockDim.x + threadIdx.x) >> 5;
  const int lane = threadIdx.x & 31;
  const int l15  = lane & 15;
  const int lhi  = lane >> 4;
  const int r0   = wave * 16;
  const int row  = r0 + l15;

  const float bet = beta[row];
  float deg = bet;
#pragma unroll
  for (int c = 0; c < SPLIT; ++c) deg += prs[c * kN + row];
  const float inv = 1.0f / deg;

  v8f acc[8];
#pragma unroll
  for (int i = 0; i < 8; ++i)
#pragma unroll
    for (int e = 0; e < 8; ++e) acc[i][e] = 0.0f;

  const float* xrow = x + (size_t)row * kFin;

#pragma unroll
  for (int kk = 0; kk < 4; ++kk) {         // K2 = 128 in steps of 32
    const int kA = kk * 32 + lhi * 16;
    float va[16];
#pragma unroll
    for (int t = 0; t < 8; ++t) {
      va[t]     = bet * xrow[kA + t];
      va[8 + t] = bet * xrow[kA + 8 + t];
    }
#pragma unroll
    for (int c = 0; c < SPLIT; ++c) {
      const float* pr = pagg + ((size_t)c * kN + row) * kFin;
#pragma unroll
      for (int t = 0; t < 8; ++t) {
        va[t]     += pr[kA + t];
        va[8 + t] += pr[kA + 8 + t];
      }
    }
    v16bf af;
#pragma unroll
    for (int t = 0; t < 16; ++t) af[t] = to_bf16(va[t] * inv);

    const bf16_t* bbase = wT + (size_t)l15 * kFin + kk * 32 + lhi * 16;
#pragma unroll
    for (int nn = 0; nn < 8; ++nn) {
      const v16bf bfrag = *(const v16bf*)(bbase + (size_t)(nn * 16) * kFin);
      acc[nn] = wmma_bf16(af, bfrag, acc[nn]);
    }
  }

#pragma unroll
  for (int nn = 0; nn < 8; ++nn) {
    const int col = nn * 16 + l15;
    const float bv = bias[col];
#pragma unroll
    for (int j = 0; j < 8; ++j)
      out[(size_t)(r0 + j + 8 * lhi) * kFout + col] = acc[nn][j] + bv;
  }
}

// ---------------------------------------------------------------------------
template <int SPLIT>
static void launch_split(const float* x, const float* adj, const float* bias,
                         const float* beta, bf16_t* xT, bf16_t* wT,
                         float* pag, float* prs, float* out, hipStream_t stream) {
  gcn_main<SPLIT><<<dim3(kStrips * SPLIT * 32 / 256), dim3(256), 0, stream>>>(adj, xT, pag, prs);
  gcn_epilogue<SPLIT><<<dim3(kStrips * 32 / 256), dim3(256), 0, stream>>>(x, bias, beta, wT, pag, prs, out);
}

extern "C" void kernel_launch(void* const* d_in, const int* in_sizes, int n_in,
                              void* d_out, int out_size, void* d_ws, size_t ws_size,
                              hipStream_t stream) {
  (void)in_sizes; (void)n_in; (void)out_size;
  const float* x    = (const float*)d_in[0];   // [8192,128]
  const float* adj  = (const float*)d_in[1];   // [8192,8192]
  const float* w    = (const float*)d_in[2];   // [128,128]
  const float* bias = (const float*)d_in[3];   // [128]
  const float* beta = (const float*)d_in[4];   // [8192]
  float* out = (float*)d_out;                  // [8192,128]

  // Per-chunk partial cost: agg f32 [8192][128] + rowsum f32 [8192]
  const size_t perChunk = (size_t)kN * kFin * 4 + (size_t)kN * 4;
  int split = 8;                               // prefer 8 for HBM saturation
  while (split > 1 && kOffVar + (size_t)split * perChunk > ws_size) split >>= 1;

  char* ws = (char*)d_ws;
  bf16_t* xT  = (bf16_t*)(ws + kOffXT);
  bf16_t* wT  = (bf16_t*)(ws + kOffWT);
  float*  pag = (float*)(ws + kOffVar);                                  // [split][8192][128]
  float*  prs = (float*)(ws + kOffVar + (size_t)split * kN * kFin * 4);  // [split][8192]

  // 1) transpose+convert x and kernel to bf16
  gcn_prep<<<dim3((kFin * kN + 255) / 256), dim3(256), 0, stream>>>(x, w, xT, wT);

  // 2) main GEMM + fused rowsum, then 3) reduce/normalize/second GEMM
  switch (split) {
    case 8: launch_split<8>(x, adj, bias, beta, xT, wT, pag, prs, out, stream); break;
    case 4: launch_split<4>(x, adj, bias, beta, xT, wT, pag, prs, out, stream); break;
    case 2: launch_split<2>(x, adj, bias, beta, xT, wT, pag, prs, out, stream); break;
    default: launch_split<1>(x, adj, bias, beta, xT, wT, pag, prs, out, stream); break;
  }
}
